// GRUSeq2SeqWithAttention_7404523618939
// MI455X (gfx1250) — matrix operand order repaired and checked by reading, not compiled, not run on hardware
//
#include <hip/hip_runtime.h>
#include <math.h>

// ---------------------------------------------------------------------------
// GRU seq2seq w/ attention for MI455X (gfx1250, wave32, WMMA bf16 16x16x32).
// Round 4: ping-pong (unroll-2) software pipeline in the 2x2 WMMA panel.
// Replaces the register-rotation (v_dual_mov chains + WMMA->VALU WAR-hazard
// v_nops seen in round 3) with alternating fragment sets: the "rotate" is now
// pure register renaming between the two unrolled halves. Everything else
// (TDM x-tile double-buffer, bf16 gate hand-off, kernel graph) unchanged.
// ---------------------------------------------------------------------------

#define B_  32
#define S_  512
#define T_  128
#define V_  32000
#define H_  512
#define H3_ 1536
#define SOS_ 1

typedef __bf16 v16bf __attribute__((ext_vector_type(16)));
typedef __bf16 v8bf  __attribute__((ext_vector_type(8)));
typedef float  v8f   __attribute__((ext_vector_type(8)));
typedef unsigned int u32x4 __attribute__((ext_vector_type(4)));
typedef int          i32x4 __attribute__((ext_vector_type(4)));
typedef int          i32x8 __attribute__((ext_vector_type(8)));

#if defined(__has_builtin)
#if __has_builtin(__builtin_amdgcn_tensor_load_to_lds)
#define HAVE_TDM 1
#endif
#if __has_builtin(__builtin_amdgcn_s_wait_tensorcnt)
#define HAVE_TCNT 1
#endif
#endif

// ---- bf16 <-> f32 (round-to-nearest-even) ----------------------------------
__device__ __forceinline__ __bf16 f2bf(float f) {
  union { float f; unsigned u; } v; v.f = f;
  unsigned r = v.u + 0x7FFFu + ((v.u >> 16) & 1u);
  unsigned short h = (unsigned short)(r >> 16);
  __bf16 o; __builtin_memcpy(&o, &h, 2); return o;
}
__device__ __forceinline__ float bf2f(__bf16 b) {
  unsigned short h; __builtin_memcpy(&h, &b, 2);
  union { unsigned u; float f; } v; v.u = ((unsigned)h) << 16;
  return v.f;
}
__device__ __forceinline__ float sigmoidf_(float x) { return 1.f / (1.f + expf(-x)); }

// ---- WMMA fragment helpers (A/B: contiguous k = 16*(lane/16) + s; D: ISA wave32 layout) ---------------
__device__ __forceinline__ v16bf load_a_rowptr(const __bf16* rowp, int k0, int lane) {
  const __bf16* p = rowp + k0 + ((lane >> 4) * 16);
  v8bf lo = *(const v8bf*)p;
  v8bf hi = *(const v8bf*)(p + 8);
  v16bf a;
#pragma unroll
  for (int i = 0; i < 8; ++i) { a[i] = lo[i]; a[i + 8] = hi[i]; }
  return a;
}
__device__ __forceinline__ v16bf load_b_frag(const __bf16* w, int ld,
                                             int n0, int k0, int lane) {
  const __bf16* p = w + (size_t)(n0 + (lane & 15)) * ld + k0 + ((lane >> 4) * 16);
  return *(const v16bf*)p;
}
__device__ __forceinline__ void store_d_f32(float* out, int ld, int m0, int n0,
                                            int lane, v8f d) {
  int col = n0 + (lane & 15), rbase = m0 + ((lane >> 4) * 8);
#pragma unroll
  for (int r = 0; r < 8; ++r) out[(size_t)(rbase + r) * ld + col] = d[r];
}
__device__ __forceinline__ void store_d_bf16(__bf16* out, int ld, int m0, int n0,
                                             int lane, v8f d) {
  int col = n0 + (lane & 15), rbase = m0 + ((lane >> 4) * 8);
#pragma unroll
  for (int r = 0; r < 8; ++r) out[(size_t)(rbase + r) * ld + col] = f2bf(d[r]);
}
__device__ __forceinline__ v8f zero8() { v8f z = {0.f,0.f,0.f,0.f,0.f,0.f,0.f,0.f}; return z; }

#define WMMA_BF16(acc, a, b) \
  acc = __builtin_amdgcn_wmma_f32_16x16x32_bf16(false, (a), false, (b), (short)0, (acc), false, false)

// ---- 2x2 register-blocked, ping-pong pipelined K=512 GEMM panel ------------
// Rows 0..15 from per-lane pointer ar0, rows 16..31 from ar1 (allows token-
// gathered embedding rows in the decoder). 16 K-slices: prologue loads slice
// 0 into set A; each loop iteration loads set B / computes set A, then loads
// set A / computes set B (no register rotation => no WMMA->VALU WAR hazard);
// epilogue consumes the last two slices.
__device__ __forceinline__ void gemm2x2(const __bf16* ar0, const __bf16* ar1,
                                        const __bf16* w, int ldw, int n0, int koff,
                                        int lane,
                                        v8f& c00, v8f& c01, v8f& c10, v8f& c11) {
  v16bf a0 = load_a_rowptr(ar0, 0, lane);
  v16bf a1 = load_a_rowptr(ar1, 0, lane);
  v16bf b0 = load_b_frag(w, ldw, n0, koff, lane);
  v16bf b1 = load_b_frag(w, ldw, n0 + 16, koff, lane);
#pragma unroll 1
  for (int k0 = 32; k0 < H_ - 64; k0 += 64) {
    v16bf d0 = load_a_rowptr(ar0, k0, lane);             // slice k0 -> set B
    v16bf d1 = load_a_rowptr(ar1, k0, lane);
    v16bf e0 = load_b_frag(w, ldw, n0, koff + k0, lane);
    v16bf e1 = load_b_frag(w, ldw, n0 + 16, koff + k0, lane);
    WMMA_BF16(c00, a0, b0); WMMA_BF16(c01, a0, b1);      // consume set A
    WMMA_BF16(c10, a1, b0); WMMA_BF16(c11, a1, b1);
    a0 = load_a_rowptr(ar0, k0 + 32, lane);              // slice k0+32 -> set A
    a1 = load_a_rowptr(ar1, k0 + 32, lane);
    b0 = load_b_frag(w, ldw, n0, koff + k0 + 32, lane);
    b1 = load_b_frag(w, ldw, n0 + 16, koff + k0 + 32, lane);
    WMMA_BF16(c00, d0, e0); WMMA_BF16(c01, d0, e1);      // consume set B
    WMMA_BF16(c10, d1, e0); WMMA_BF16(c11, d1, e1);
  }
  {                                                       // slices H-64, H-32
    v16bf d0 = load_a_rowptr(ar0, H_ - 32, lane);
    v16bf d1 = load_a_rowptr(ar1, H_ - 32, lane);
    v16bf e0 = load_b_frag(w, ldw, n0, koff + H_ - 32, lane);
    v16bf e1 = load_b_frag(w, ldw, n0 + 16, koff + H_ - 32, lane);
    WMMA_BF16(c00, a0, b0); WMMA_BF16(c01, a0, b1);
    WMMA_BF16(c10, a1, b0); WMMA_BF16(c11, a1, b1);
    WMMA_BF16(c00, d0, e0); WMMA_BF16(c01, d0, e1);
    WMMA_BF16(c10, d1, e0); WMMA_BF16(c11, d1, e1);
  }
}

// ---- TDM staging: async DMA of one contiguous 32KB bf16 tile into LDS ------
__device__ __forceinline__ void stage_x(__bf16* dst_lds, const __bf16* src_global,
                                        unsigned lds_byte_off, int tid, int wave) {
#if defined(HAVE_TDM)
  if (wave == 0) {
    unsigned long long ga = (unsigned long long)(uintptr_t)src_global;
    u32x4 g0;
    g0[0] = 1u;                                                // count=1
    g0[1] = lds_byte_off;                                      // lds_addr
    g0[2] = (unsigned)(ga & 0xFFFFFFFFu);                      // global_addr lo
    g0[3] = (unsigned)((ga >> 32) & 0x01FFFFFFu) | (2u << 30); // addr hi | type=2
    i32x8 g1;
    g1[0] = (int)(1u << 16);                  // data_size = 2 bytes
    g1[1] = (int)((16384u & 0xFFFFu) << 16);  // tensor_dim0[15:0]
    g1[2] = (int)(1u << 16);                  // tensor_dim0 hi | tensor_dim1=1
    g1[3] = (int)((16384u & 0xFFFFu) << 16);  // tensor_dim1 hi | tile_dim0
    g1[4] = 1;                                // tile_dim1=1, tile_dim2=0
    g1[5] = 16384;                            // tensor_dim0_stride lo
    g1[6] = 0; g1[7] = 0;
    i32x4 g2 = {0, 0, 0, 0}, g3 = {0, 0, 0, 0};
#if __clang_major__ >= 23
    i32x8 g4 = {0, 0, 0, 0, 0, 0, 0, 0};
    __builtin_amdgcn_tensor_load_to_lds(g0, g1, g2, g3, g4, 0);
#else
    __builtin_amdgcn_tensor_load_to_lds(g0, g1, g2, g3, 0);
#endif
  }
#else
  for (int i = tid; i < B_ * H_; i += 1024) dst_lds[i] = src_global[i];
#endif
}
__device__ __forceinline__ void wait_x(int wave) {
#if defined(HAVE_TDM) && defined(HAVE_TCNT)
  if (wave == 0) __builtin_amdgcn_s_wait_tensorcnt(0);
#endif
}

// ======================= preprocessing kernels ==============================
__global__ void cast_bf16_kernel(__bf16* __restrict__ dst, const float* __restrict__ src, size_t n) {
  size_t i = (size_t)blockIdx.x * blockDim.x + threadIdx.x;
  size_t st = (size_t)gridDim.x * blockDim.x;
  for (; i < n; i += st) dst[i] = f2bf(src[i]);
}
__global__ void cast_cat_kernel(__bf16* __restrict__ dst,
                                const float* __restrict__ Wih,
                                const float* __restrict__ Whh) {
  size_t n = (size_t)H3_ * 1024;
  size_t i = (size_t)blockIdx.x * blockDim.x + threadIdx.x;
  size_t st = (size_t)gridDim.x * blockDim.x;
  for (; i < n; i += st) {
    int row = (int)(i >> 10), k = (int)(i & 1023);
    float v = (k < H_) ? Wih[(size_t)row * H_ + k] : Whh[(size_t)row * H_ + (k - H_)];
    dst[i] = f2bf(v);
  }
}
__global__ void gather_xs_kernel(__bf16* __restrict__ xs, const __bf16* __restrict__ emb_bf,
                                 const int* __restrict__ input_seq) {
  size_t n = (size_t)S_ * B_ * H_;
  size_t i = (size_t)blockIdx.x * blockDim.x + threadIdx.x;
  size_t st = (size_t)gridDim.x * blockDim.x;
  for (; i < n; i += st) {
    int h = (int)(i & (H_ - 1));
    int sb = (int)(i >> 9);
    int b = sb & (B_ - 1);
    int s = sb >> 5;
    xs[i] = emb_bf[(size_t)input_seq[(size_t)b * S_ + s] * H_ + h];
  }
}

// ======================= encoder persistent scan ============================
__global__ __launch_bounds__(1024) void encoder_scan_kernel(
    const __bf16* __restrict__ xs,     // [S][B][H]
    const __bf16* __restrict__ WcatF,  // [1536][1024] = [Wih|Whh]
    const __bf16* __restrict__ WcatB,
    const float* bihF, const float* bhhF,
    const float* bihB, const float* bhhB,
    float* __restrict__ encF,          // [B][S][H]
    float* __restrict__ encB,
    float* __restrict__ hlast)         // [2][B][H]
{
  extern __shared__ char smem[];
  __bf16* xbuf = (__bf16*)smem;                       // 2 x 32KB, offset 0
  __bf16* grz  = xbuf + 2 * B_ * H_;                  // [32][1024]
  __bf16* gni  = grz + B_ * 1024;                     // [32][512]
  __bf16* gnh  = gni + B_ * H_;                       // [32][512]
  __bf16* hbf  = gnh + B_ * H_;                       // [32][512]

  const int dir = blockIdx.x;
  const __bf16* Wcat = dir ? WcatB : WcatF;
  const float* bih = dir ? bihB : bihF;
  const float* bhh = dir ? bhhB : bhhF;
  float* enc = dir ? encB : encF;

  const int tid = threadIdx.x, lane = tid & 31, wave = tid >> 5;
  const int lrow = lane & 15;

  for (int i = tid; i < B_ * H_; i += 1024) hbf[i] = f2bf(0.f);
  {
    const int s0 = dir ? (S_ - 1) : 0;
    stage_x(xbuf, xs + (size_t)s0 * B_ * H_, 0u, tid, wave);
    wait_x(wave);
  }
  __syncthreads();

  const __bf16* hr0 = hbf + (size_t)lrow * H_;
  const __bf16* hr1 = hbf + (size_t)(16 + lrow) * H_;

  for (int t = 0; t < S_; ++t) {
    const int s = dir ? (S_ - 1 - t) : t;
    const __bf16* xcur = xbuf + (t & 1) * (B_ * H_);
    const __bf16* xr0 = xcur + (size_t)lrow * H_;
    const __bf16* xr1 = xcur + (size_t)(16 + lrow) * H_;

    if (t + 1 < S_) {   // async DMA of next timestep's x tile
      const int sn = dir ? (S_ - 1 - (t + 1)) : (t + 1);
      stage_x(xbuf + ((t + 1) & 1) * (B_ * H_), xs + (size_t)sn * B_ * H_,
              (unsigned)(((t + 1) & 1) * (B_ * H_ * 2)), tid, wave);
    }

#pragma unroll 1
    for (int i = 0; i < 2; ++i) {
      int mm = wave * 2 + i;                  // [0,64) macro-tiles
      v8f c00 = zero8(), c01 = zero8(), c10 = zero8(), c11 = zero8();
      __bf16* dst; int dld, dn0;
      if (mm < 32) {                          // r,z gates: K = 1024
        int n0 = mm * 32;
        gemm2x2(xr0, xr1, Wcat, 1024, n0, 0, lane, c00, c01, c10, c11);
        gemm2x2(hr0, hr1, Wcat, 1024, n0, H_, lane, c00, c01, c10, c11);
        dst = grz; dld = 1024; dn0 = n0;
      } else if (mm < 48) {                   // n-gate input part: K = 512
        int n0 = 1024 + (mm - 32) * 32;
        gemm2x2(xr0, xr1, Wcat, 1024, n0, 0, lane, c00, c01, c10, c11);
        dst = gni; dld = H_; dn0 = n0 - 1024;
      } else {                                // n-gate hidden part: K = 512
        int n0 = 1024 + (mm - 48) * 32;
        gemm2x2(hr0, hr1, Wcat, 1024, n0, H_, lane, c00, c01, c10, c11);
        dst = gnh; dld = H_; dn0 = n0 - 1024;
      }
      store_d_bf16(dst, dld, 0,  dn0,      lane, c00);
      store_d_bf16(dst, dld, 0,  dn0 + 16, lane, c01);
      store_d_bf16(dst, dld, 16, dn0,      lane, c10);
      store_d_bf16(dst, dld, 16, dn0 + 16, lane, c11);
    }
    __syncthreads();

    for (int e = tid; e < B_ * H_; e += 1024) {
      int b = e >> 9, j = e & (H_ - 1);
      float r = sigmoidf_(bf2f(grz[b * 1024 + j]) + bih[j] + bhh[j]);
      float z = sigmoidf_(bf2f(grz[b * 1024 + H_ + j]) + bih[H_ + j] + bhh[H_ + j]);
      float n = tanhf(bf2f(gni[e]) + bih[1024 + j] + r * (bf2f(gnh[e]) + bhh[1024 + j]));
      float hold = bf2f(hbf[e]);
      float hnew = (1.f - z) * n + z * hold;
      enc[((size_t)b * S_ + s) * H_ + j] = hnew;
      if (t == S_ - 1) hlast[((size_t)dir * B_ + b) * H_ + j] = hnew;
      hbf[e] = f2bf(hnew);
    }
    wait_x(wave);
    __syncthreads();
  }
}

__global__ void merge_enc_kernel(float* __restrict__ encF, const float* __restrict__ encB) {
  size_t n = (size_t)B_ * S_ * H_;
  size_t i = (size_t)blockIdx.x * blockDim.x + threadIdx.x;
  size_t st = (size_t)gridDim.x * blockDim.x;
  for (; i < n; i += st) encF[i] += encB[i];
}

__global__ void init_dec_kernel(const float* __restrict__ hlast, __bf16* __restrict__ hstate_bf,
                                int* __restrict__ tok, float* __restrict__ loss) {
  int i = blockIdx.x * blockDim.x + threadIdx.x;
  if (i < B_ * H_) hstate_bf[i] = f2bf(hlast[i] + hlast[B_ * H_ + i]);
  if (i < B_) { tok[i] = SOS_; loss[i] = 0.f; }
}

// ======================= decoder step kernels ===============================
__global__ __launch_bounds__(1024) void dec_gru_kernel(
    const __bf16* __restrict__ emb_bf, const int* __restrict__ tok,
    __bf16* __restrict__ hstate_bf,       // [32][512] read as h, rewritten as hn
    const __bf16* __restrict__ WcatD,     // [1536][1024]
    const float* bih, const float* bhh,
    float* __restrict__ hn_out)           // [32][512] f32
{
  extern __shared__ char smem[];
  float* grz = (float*)smem;
  float* gni = grz + B_ * 1024;
  float* gnh = gni + B_ * H_;
  const int tid = threadIdx.x, lane = tid & 31, wave = tid >> 5;
  const int lrow = lane & 15;

  const __bf16* xr0 = emb_bf + (size_t)tok[lrow] * H_;        // gathered rows
  const __bf16* xr1 = emb_bf + (size_t)tok[16 + lrow] * H_;
  const __bf16* hr0 = hstate_bf + (size_t)lrow * H_;
  const __bf16* hr1 = hstate_bf + (size_t)(16 + lrow) * H_;

#pragma unroll 1
  for (int i = 0; i < 2; ++i) {
    int mm = wave * 2 + i;
    v8f c00 = zero8(), c01 = zero8(), c10 = zero8(), c11 = zero8();
    float* dst; int dld, dn0;
    if (mm < 32) {
      int n0 = mm * 32;
      gemm2x2(xr0, xr1, WcatD, 1024, n0, 0, lane, c00, c01, c10, c11);
      gemm2x2(hr0, hr1, WcatD, 1024, n0, H_, lane, c00, c01, c10, c11);
      dst = grz; dld = 1024; dn0 = n0;
    } else if (mm < 48) {
      int n0 = 1024 + (mm - 32) * 32;
      gemm2x2(xr0, xr1, WcatD, 1024, n0, 0, lane, c00, c01, c10, c11);
      dst = gni; dld = H_; dn0 = n0 - 1024;
    } else {
      int n0 = 1024 + (mm - 48) * 32;
      gemm2x2(hr0, hr1, WcatD, 1024, n0, H_, lane, c00, c01, c10, c11);
      dst = gnh; dld = H_; dn0 = n0 - 1024;
    }
    store_d_f32(dst, dld, 0,  dn0,      lane, c00);
    store_d_f32(dst, dld, 0,  dn0 + 16, lane, c01);
    store_d_f32(dst, dld, 16, dn0,      lane, c10);
    store_d_f32(dst, dld, 16, dn0 + 16, lane, c11);
  }
  __syncthreads();
  for (int e = tid; e < B_ * H_; e += 1024) {
    int b = e >> 9, j = e & (H_ - 1);
    float r = sigmoidf_(grz[b * 1024 + j] + bih[j] + bhh[j]);
    float z = sigmoidf_(grz[b * 1024 + H_ + j] + bih[H_ + j] + bhh[H_ + j]);
    float n = tanhf(gni[e] + bih[1024 + j] + r * (gnh[e] + bhh[1024 + j]));
    float hold = bf2f(hstate_bf[e]);
    float hnew = (1.f - z) * n + z * hold;
    hn_out[e] = hnew;
    hstate_bf[e] = f2bf(hnew);
  }
}

// D2: attention: scores->softmax->ctx; writes a_att = [ctx | hn] (bf16).
__global__ __launch_bounds__(256) void attention_kernel(
    const float* __restrict__ hn, const float* __restrict__ enc, __bf16* __restrict__ a_att)
{
  __shared__ float sh_h[H_];
  __shared__ float sc[S_];
  __shared__ float red[256];
  const int b = blockIdx.x, t = threadIdx.x;
  for (int i = t; i < H_; i += 256) sh_h[i] = hn[(size_t)b * H_ + i];
  __syncthreads();
  const float* erow = enc + (size_t)b * S_ * H_;
  for (int s = t; s < S_; s += 256) {
    float acc = 0.f;
    for (int k = 0; k < H_; ++k) acc += sh_h[k] * erow[(size_t)s * H_ + k];
    sc[s] = acc;
  }
  __syncthreads();
  float m = -INFINITY;
  for (int s = t; s < S_; s += 256) m = fmaxf(m, sc[s]);
  red[t] = m; __syncthreads();
  for (int o = 128; o > 0; o >>= 1) { if (t < o) red[t] = fmaxf(red[t], red[t + o]); __syncthreads(); }
  m = red[0]; __syncthreads();
  float sum = 0.f;
  for (int s = t; s < S_; s += 256) { float w = expf(sc[s] - m); sc[s] = w; sum += w; }
  red[t] = sum; __syncthreads();
  for (int o = 128; o > 0; o >>= 1) { if (t < o) red[t] += red[t + o]; __syncthreads(); }
  float inv = 1.f / red[0];
  __syncthreads();
  for (int h = t; h < H_; h += 256) {
    float acc = 0.f;
    for (int s = 0; s < S_; ++s) acc += sc[s] * erow[(size_t)s * H_ + h];
    a_att[(size_t)b * 1024 + h] = f2bf(acc * inv);
  }
  for (int h = t; h < H_; h += 256) a_att[(size_t)b * 1024 + H_ + h] = f2bf(sh_h[h]);
}

// D3: att_out = tanh([ctx,hn] @ att_W^T + att_b). 16 macro-tiles, K=1024.
__global__ __launch_bounds__(1024) void attproj_kernel(
    const __bf16* __restrict__ a_att, const __bf16* __restrict__ attW,
    const float* __restrict__ att_b, __bf16* __restrict__ att_out)
{
  const int lane = threadIdx.x & 31, wave = threadIdx.x >> 5;
  if (wave >= 16) return;                   // 16 macro-tiles of 32 output cols
  const int lrow = lane & 15;
  const __bf16* ar0 = a_att + (size_t)lrow * 1024;
  const __bf16* ar1 = a_att + (size_t)(16 + lrow) * 1024;
  int n0 = wave * 32;
  v8f c00 = zero8(), c01 = zero8(), c10 = zero8(), c11 = zero8();
  gemm2x2(ar0, ar1, attW, 1024, n0, 0, lane, c00, c01, c10, c11);    // K 0..511
  gemm2x2(ar0 + H_, ar1 + H_, attW, 1024, n0, H_, lane, c00, c01, c10, c11);   // K 512..1023
  int col = n0 + lrow, rb = (lane >> 4) * 8;
  float bia0 = att_b[col], bia1 = att_b[col + 16];
#pragma unroll
  for (int r = 0; r < 8; ++r) {
    att_out[(size_t)(rb + r) * H_ + col]           = f2bf(tanhf(c00[r] + bia0));
    att_out[(size_t)(rb + r) * H_ + col + 16]      = f2bf(tanhf(c01[r] + bia1));
    att_out[(size_t)(16 + rb + r) * H_ + col]      = f2bf(tanhf(c10[r] + bia0));
    att_out[(size_t)(16 + rb + r) * H_ + col + 16] = f2bf(tanhf(c11[r] + bia1));
  }
}

// D4: logits = att_out @ out_W^T + out_b. 1000 macro-tiles = 125 blks x 8 wv.
__global__ __launch_bounds__(256) void logits_kernel(
    const __bf16* __restrict__ att_out, const __bf16* __restrict__ outW,
    const float* __restrict__ out_b, float* __restrict__ logits)
{
  const int lane = threadIdx.x & 31, wave = threadIdx.x >> 5;
  const int lrow = lane & 15;
  int mm = blockIdx.x * 8 + wave;           // [0,1000)
  int n0 = mm * 32;
  __builtin_prefetch(outW + (size_t)(n0 + lrow) * H_, 0, 1);  // global_prefetch_b8
  const __bf16* ar0 = att_out + (size_t)lrow * H_;
  const __bf16* ar1 = att_out + (size_t)(16 + lrow) * H_;
  v8f c00 = zero8(), c01 = zero8(), c10 = zero8(), c11 = zero8();
  gemm2x2(ar0, ar1, outW, H_, n0, 0, lane, c00, c01, c10, c11);
  int col = n0 + lrow, rb = (lane >> 4) * 8;
  float bia0 = out_b[col], bia1 = out_b[col + 16];
#pragma unroll
  for (int r = 0; r < 8; ++r) {
    logits[(size_t)(rb + r) * V_ + col]           = c00[r] + bia0;
    logits[(size_t)(rb + r) * V_ + col + 16]      = c01[r] + bia1;
    logits[(size_t)(16 + rb + r) * V_ + col]      = c10[r] + bia0;
    logits[(size_t)(16 + rb + r) * V_ + col + 16] = c11[r] + bia1;
  }
}

// D5: per-row (b) max/argmax + logsumexp + NLL accumulation + next token.
__global__ __launch_bounds__(256) void vocab_reduce_kernel(
    const float* __restrict__ logits, const int* __restrict__ target_seq, int t,
    int* __restrict__ tok, float* __restrict__ loss)
{
  __shared__ float rmax[256]; __shared__ int rarg[256]; __shared__ float rsum[256];
  const int b = blockIdx.x, tid = threadIdx.x;
  const float* lr = logits + (size_t)b * V_;
  float m = -INFINITY; int arg = 0;
  for (int v = tid; v < V_; v += 256) { float x = lr[v]; if (x > m) { m = x; arg = v; } }
  rmax[tid] = m; rarg[tid] = arg; __syncthreads();
  for (int o = 128; o > 0; o >>= 1) {
    if (tid < o && rmax[tid + o] > rmax[tid]) { rmax[tid] = rmax[tid + o]; rarg[tid] = rarg[tid + o]; }
    __syncthreads();
  }
  m = rmax[0];
  float s = 0.f;
  for (int v = tid; v < V_; v += 256) s += expf(lr[v] - m);
  rsum[tid] = s; __syncthreads();
  for (int o = 128; o > 0; o >>= 1) { if (tid < o) rsum[tid] += rsum[tid + o]; __syncthreads(); }
  if (tid == 0) {
    int tgt = target_seq[(size_t)b * T_ + t];
    loss[b] += -(lr[tgt] - m - logf(rsum[0]));
    tok[b] = rarg[0];
  }
}

__global__ void finalize_kernel(const float* __restrict__ loss, float* __restrict__ out) {
  if (threadIdx.x == 0) {
    float s = 0.f;
    for (int b = 0; b < B_; ++b) s += loss[b];
    out[0] = s / ((float)B_ * (float)T_);
  }
}

// ======================= host-side orchestration ============================
extern "C" void kernel_launch(void* const* d_in, const int* in_sizes, int n_in,
                              void* d_out, int out_size, void* d_ws, size_t ws_size,
                              hipStream_t stream) {
  (void)in_sizes; (void)n_in; (void)out_size; (void)ws_size;
  const int*   input_seq  = (const int*)d_in[0];
  const int*   target_seq = (const int*)d_in[1];
  const float* emb    = (const float*)d_in[2];
  const float* ef_Wih = (const float*)d_in[3];
  const float* ef_Whh = (const float*)d_in[4];
  const float* ef_bih = (const float*)d_in[5];
  const float* ef_bhh = (const float*)d_in[6];
  const float* eb_Wih = (const float*)d_in[7];
  const float* eb_Whh = (const float*)d_in[8];
  const float* eb_bih = (const float*)d_in[9];
  const float* eb_bhh = (const float*)d_in[10];
  const float* d_Wih  = (const float*)d_in[11];
  const float* d_Whh  = (const float*)d_in[12];
  const float* d_bih  = (const float*)d_in[13];
  const float* d_bhh  = (const float*)d_in[14];
  const float* att_W  = (const float*)d_in[15];
  const float* att_b  = (const float*)d_in[16];
  const float* out_W  = (const float*)d_in[17];
  const float* out_b  = (const float*)d_in[18];
  float* out = (float*)d_out;

  char* w = (char*)d_ws;
  auto alloc = [&](size_t bytes) -> void* {
    void* p = (void*)w; w += ((bytes + 255) & ~(size_t)255); return p;
  };
  __bf16* emb_bf   = (__bf16*)alloc((size_t)V_ * H_ * 2);
  __bf16* outW_bf  = (__bf16*)alloc((size_t)V_ * H_ * 2);
  __bf16* attW_bf  = (__bf16*)alloc((size_t)H_ * 1024 * 2);
  __bf16* WcatF    = (__bf16*)alloc((size_t)H3_ * 1024 * 2);
  __bf16* WcatB    = (__bf16*)alloc((size_t)H3_ * 1024 * 2);
  __bf16* WcatD    = (__bf16*)alloc((size_t)H3_ * 1024 * 2);
  __bf16* xs_bf    = (__bf16*)alloc((size_t)S_ * B_ * H_ * 2);
  float*  encF     = (float*) alloc((size_t)B_ * S_ * H_ * 4);
  float*  encB     = (float*) alloc((size_t)B_ * S_ * H_ * 4);
  float*  hlast    = (float*) alloc((size_t)2 * B_ * H_ * 4);
  __bf16* hstate   = (__bf16*)alloc((size_t)B_ * H_ * 2);
  float*  hn_f32   = (float*) alloc((size_t)B_ * H_ * 4);
  __bf16* a_att    = (__bf16*)alloc((size_t)B_ * 1024 * 2);
  __bf16* att_out  = (__bf16*)alloc((size_t)B_ * H_ * 2);
  float*  logits   = (float*) alloc((size_t)B_ * V_ * 4);
  int*    tok      = (int*)   alloc((size_t)B_ * 4);
  float*  loss     = (float*) alloc((size_t)B_ * 4);

  cast_bf16_kernel<<<2048, 256, 0, stream>>>(emb_bf, emb, (size_t)V_ * H_);
  cast_bf16_kernel<<<2048, 256, 0, stream>>>(outW_bf, out_W, (size_t)V_ * H_);
  cast_bf16_kernel<<<512, 256, 0, stream>>>(attW_bf, att_W, (size_t)H_ * 1024);
  cast_cat_kernel<<<1024, 256, 0, stream>>>(WcatF, ef_Wih, ef_Whh);
  cast_cat_kernel<<<1024, 256, 0, stream>>>(WcatB, eb_Wih, eb_Whh);
  cast_cat_kernel<<<1024, 256, 0, stream>>>(WcatD, d_Wih, d_Whh);
  gather_xs_kernel<<<2048, 256, 0, stream>>>(xs_bf, emb_bf, input_seq);

  const size_t enc_lds = (size_t)(2 * B_ * H_ + B_ * 1024 + 3 * B_ * H_) * 2; // 229376 B
  encoder_scan_kernel<<<2, 1024, enc_lds, stream>>>(
      xs_bf, WcatF, WcatB, ef_bih, ef_bhh, eb_bih, eb_bhh, encF, encB, hlast);
  merge_enc_kernel<<<4096, 1024, 0, stream>>>(encF, encB);
  init_dec_kernel<<<(B_ * H_ + 255) / 256, 256, 0, stream>>>(hlast, hstate, tok, loss);

  const size_t gru_lds = (size_t)(B_ * 1024 + 2 * B_ * H_) * 4; // 262144 B
  for (int t = 0; t < T_; ++t) {
    dec_gru_kernel<<<1, 1024, gru_lds, stream>>>(emb_bf, tok, hstate, WcatD, d_bih, d_bhh, hn_f32);
    attention_kernel<<<B_, 256, 0, stream>>>(hn_f32, encF, a_att);
    attproj_kernel<<<1, 1024, 0, stream>>>(a_att, attW_bf, att_b, att_out);
    logits_kernel<<<125, 256, 0, stream>>>(att_out, outW_bf, out_b, logits);
    vocab_reduce_kernel<<<B_, 256, 0, stream>>>(logits, target_seq, t, tok, loss);
  }
  finalize_kernel<<<1, 32, 0, stream>>>(loss, out);
}
